// GAT_layer_17514876634214
// MI455X (gfx1250) — matrix operand order repaired and checked by reading, not compiled, not run on hardware
//
#include <hip/hip_runtime.h>
#include <hip/hip_bf16.h>

#define D_IN 128
#define HC   128
#define NHEAD 4
#define NEG_SLOPE 0.2f
#define LN_EPS 1e-5
#define BM 64              // rows of x per block in the WMMA GEMM

typedef __attribute__((ext_vector_type(16))) _Float16 v16h;
typedef __attribute__((ext_vector_type(8)))  _Float16 v8h;
typedef __attribute__((ext_vector_type(8)))  float    v8f;

// ---------- order-preserving float <-> uint for integer atomicMax ----------
__device__ __forceinline__ unsigned fenc(float f) {
    unsigned u = __float_as_uint(f);
    return (u & 0x80000000u) ? ~u : (u | 0x80000000u);
}
__device__ __forceinline__ float fdec(unsigned u) {
    return __uint_as_float((u & 0x80000000u) ? (u & 0x7FFFFFFFu) : ~u);
}

// ---------- 0. convert weights to f16 (row-major [K=128][N=128], wl then wr) ----------
__global__ __launch_bounds__(256) void k_convert_w(const float* __restrict__ wl,
                                                   const float* __restrict__ wr,
                                                   _Float16* __restrict__ w16) {
    int i = blockIdx.x * 256 + threadIdx.x;
    if (i < D_IN * HC) {
        w16[i]             = (_Float16)wl[i];
        w16[D_IN * HC + i] = (_Float16)wr[i];
    }
}

// ---------- 1. WMMA GEMM: xl = x@w_l + b_l ; xr = x@w_r + b_r ----------
// 512 threads = 16 waves. wave[3] selects {w_l, w_r}; wave[2:0] selects the N-tile.
// Block covers BM=64 rows: x-tile staged once to LDS as f16 (rows padded to 136
// halfs -> 272B stride -> bank stride 4 -> conflict-free b128 A-fragment reads).
// Each wave keeps its 4 B K-chunks resident and runs 4 M-tiles x 4 K-chunks = 16 WMMAs.
__global__ __launch_bounds__(512) void k_gemm(const float* __restrict__ x,
                                              const _Float16* __restrict__ w16,
                                              const float* __restrict__ b_l,
                                              const float* __restrict__ b_r,
                                              float* __restrict__ xl,
                                              float* __restrict__ xr,
                                              int nrows) {
    __shared__ _Float16 lds[BM][136];

    const int row0 = blockIdx.x * BM;

    // cooperative stage+convert: BM*128 f32 = 8192 elems = 2048 float4, 512 threads * 4
    #pragma unroll
    for (int it = 0; it < 4; ++it) {
        int idx = it * 512 + threadIdx.x;     // float4 index
        int r   = idx >> 5;                   // 32 float4 per row
        int c4  = idx & 31;
        int gr  = row0 + r;
        int grs = (gr < nrows) ? gr : 0;      // clamp; garbage rows never stored
        float4 v = ((const float4*)(x + (size_t)grs * D_IN))[c4];
        _Float16* p = &lds[r][c4 * 4];
        p[0] = (_Float16)v.x; p[1] = (_Float16)v.y;
        p[2] = (_Float16)v.z; p[3] = (_Float16)v.w;
    }
    __syncthreads();

    const int wave  = threadIdx.x >> 5;
    const int lane  = threadIdx.x & 31;
    const int which = wave >> 3;              // 0 -> l, 1 -> r
    const int n0    = (wave & 7) * 16;        // N-tile origin

    const _Float16* w    = w16 + which * (D_IN * HC);
    const float*    bias = which ? b_r : b_l;
    float*          out  = which ? xr : xl;

    // B fragments (32x16 f16): lane l holds col n0+(l&15), K in table order (same as A); loaded once
    v16h bfr[4];
    #pragma unroll
    for (int kc = 0; kc < 4; ++kc)
        #pragma unroll
        for (int s = 0; s < 16; ++s) {
            int k = kc * 32 + ((s < 8) ? (lane >> 4) * 8 + s : 16 + (lane >> 4) * 8 + (s - 8));
            bfr[kc][s] = w[(size_t)k * HC + n0 + (lane & 15)];
        }

    const int mlo = lane & 15;
    const int hi  = lane >> 4;
    const int n   = n0 + mlo;
    const float bn = bias[n];

    #pragma unroll
    for (int mt = 0; mt < BM / 16; ++mt) {
        v8f acc = {};
        #pragma unroll
        for (int kc = 0; kc < 4; ++kc) {
            const int kk = kc * 32;
            // A fragment (16x32 f16): lane<16 -> row mlo, K = kk+[0..7] & kk+16+[0..7];
            // lane>=16 -> K = kk+8+[0..7] & kk+24+[0..7]  (ISA 16-bit A layout)
            const _Float16* ap = &lds[mt * 16 + mlo][kk + hi * 8];
            v8h a0 = *(const v8h*)ap;          // ds_load_b128
            v8h a1 = *(const v8h*)(ap + 16);   // ds_load_b128
            v16h a;
            #pragma unroll
            for (int j = 0; j < 8; ++j) { a[j] = a0[j]; a[8 + j] = a1[j]; }
            acc = __builtin_amdgcn_wmma_f32_16x16x32_f16(false, a, false, bfr[kc],
                                                         (short)0, acc, false, false);
        }
        // D layout: VGPR r -> M = hi*8 + r, N = n0 + (lane&15)
        const int mbase = row0 + mt * 16 + hi * 8;
        #pragma unroll
        for (int r = 0; r < 8; ++r) {
            int mrow = mbase + r;
            if (mrow < nrows) out[(size_t)mrow * HC + n] = acc[r] + bn;
        }
    }
}

// ---------- 2. init: zero d_out / denom / acc, floor mmax ----------
__global__ __launch_bounds__(256) void k_init(float* __restrict__ out, int total,
                                              float* __restrict__ denom,
                                              unsigned* __restrict__ mmax, int nh,
                                              double* __restrict__ acc) {
    int i = blockIdx.x * 256 + threadIdx.x;
    if (i < total) out[i] = 0.0f;
    if (i < nh) { denom[i] = 0.0f; mmax[i] = 0u; }   // 0u decodes below every real float
    if (i < 2) acc[i] = 0.0;
}

// ---------- 3. per-edge logits + segment max (one wave per edge) ----------
__global__ __launch_bounds__(256) void k_edge_logits(const int* __restrict__ ei, int E, int N,
                                                     const float* __restrict__ xl,
                                                     const float* __restrict__ xr,
                                                     const float* __restrict__ att,
                                                     float* __restrict__ logits,
                                                     unsigned* __restrict__ mmax) {
    const int e    = blockIdx.x * 8 + (threadIdx.x >> 5);
    const int lane = threadIdx.x & 31;
    const int Et   = E + N;
    if (e >= Et) return;
    int src, dst;
    if (e < E) { src = ei[e]; dst = ei[E + e]; } else { src = dst = e - E; }

    const float4 a4 = ((const float4*)att)[lane];                       // att flat [H*C]
    const float4 l4 = ((const float4*)(xl + (size_t)src * HC))[lane];
    const float4 r4 = ((const float4*)(xr + (size_t)dst * HC))[lane];

    float v, s = 0.0f;
    v = l4.x + r4.x; v = v > 0.0f ? v : v * NEG_SLOPE; s += v * a4.x;
    v = l4.y + r4.y; v = v > 0.0f ? v : v * NEG_SLOPE; s += v * a4.y;
    v = l4.z + r4.z; v = v > 0.0f ? v : v * NEG_SLOPE; s += v * a4.z;
    v = l4.w + r4.w; v = v > 0.0f ? v : v * NEG_SLOPE; s += v * a4.w;

    // reduce the 8 lanes belonging to one head (head = lane>>3)
    s += __shfl_xor(s, 1);
    s += __shfl_xor(s, 2);
    s += __shfl_xor(s, 4);
    if ((lane & 7) == 0) {
        int h = lane >> 3;
        logits[(size_t)e * NHEAD + h] = s;
        atomicMax(mmax + (size_t)dst * NHEAD + h, fenc(s));
    }
}

// ---------- 4. exp(logit - max) + segment sum (one thread per edge*head) ----------
__global__ __launch_bounds__(256) void k_edge_exp(const int* __restrict__ ei, int E, int N,
                                                  const unsigned* __restrict__ mmax,
                                                  float* __restrict__ logits,   // in: logit, out: exp
                                                  float* __restrict__ denom) {
    const int Et = E + N;
    int i = blockIdx.x * 256 + threadIdx.x;
    if (i >= Et * NHEAD) return;
    int e = i >> 2, h = i & 3;
    int dst = (e < E) ? ei[E + e] : (e - E);
    float m  = fdec(mmax[(size_t)dst * NHEAD + h]);
    float ex = expf(logits[i] - m);
    logits[i] = ex;
    unsafeAtomicAdd(denom + (size_t)dst * NHEAD + h, ex);
}

// ---------- 5. alpha-weighted scatter-add of messages (one wave per edge) ----------
__global__ __launch_bounds__(256) void k_edge_agg(const int* __restrict__ ei, int E, int N,
                                                  const float* __restrict__ xl,
                                                  const float* __restrict__ ex,
                                                  const float* __restrict__ denom,
                                                  float* __restrict__ out) {
    const int e    = blockIdx.x * 8 + (threadIdx.x >> 5);
    const int lane = threadIdx.x & 31;
    const int Et   = E + N;
    if (e >= Et) return;
    int src, dst;
    if (e < E) { src = ei[e]; dst = ei[E + e]; } else { src = dst = e - E; }

    const int h = lane >> 3;
    const float alpha = ex[(size_t)e * NHEAD + h] / denom[(size_t)dst * NHEAD + h];
    const float4 l4 = ((const float4*)(xl + (size_t)src * HC))[lane];
    float* op = out + (size_t)dst * HC + lane * 4;
    unsafeAtomicAdd(op + 0, alpha * l4.x);
    unsafeAtomicAdd(op + 1, alpha * l4.y);
    unsafeAtomicAdd(op + 2, alpha * l4.z);
    unsafeAtomicAdd(op + 3, alpha * l4.w);
}

// ---------- 6. add bias, accumulate global sum / sumsq ----------
__global__ __launch_bounds__(256) void k_bias_stats(float* __restrict__ out,
                                                    const float* __restrict__ bias, int total,
                                                    double* __restrict__ acc) {
    int i = blockIdx.x * 256 + threadIdx.x;
    float y = 0.0f;
    if (i < total) {
        y = out[i] + bias[i & (HC - 1)];
        out[i] = y;
    }
    float s = y, q = y * y;
    #pragma unroll
    for (int o = 16; o; o >>= 1) { s += __shfl_xor(s, o); q += __shfl_xor(q, o); }
    __shared__ float bs, bq;
    if (threadIdx.x == 0) { bs = 0.0f; bq = 0.0f; }
    __syncthreads();
    if ((threadIdx.x & 31) == 0) { atomicAdd(&bs, s); atomicAdd(&bq, q); }   // ds_add_f32
    __syncthreads();
    if (threadIdx.x == 0) {
        unsafeAtomicAdd(acc + 0, (double)bs);
        unsafeAtomicAdd(acc + 1, (double)bq);
    }
}

// ---------- 7. finalize mean / inv-std ----------
__global__ void k_finalize(const double* __restrict__ acc, float* __restrict__ mi, double total) {
    double mean = acc[0] / total;
    double var  = acc[1] / total - mean * mean;
    if (var < 0.0) var = 0.0;
    mi[0] = (float)mean;
    mi[1] = (float)(1.0 / (sqrt(var) + LN_EPS));
}

// ---------- 8. apply graph-LN affine ----------
__global__ __launch_bounds__(256) void k_apply(float* __restrict__ out,
                                               const float* __restrict__ mi,
                                               const float* __restrict__ ln_w,
                                               const float* __restrict__ ln_b, int total) {
    int i = blockIdx.x * 256 + threadIdx.x;
    if (i < total) {
        int c = i & (HC - 1);
        out[i] = (out[i] - mi[0]) * mi[1] * ln_w[c] + ln_b[c];
    }
}

extern "C" void kernel_launch(void* const* d_in, const int* in_sizes, int n_in,
                              void* d_out, int out_size, void* d_ws, size_t ws_size,
                              hipStream_t stream) {
    const float* x    = (const float*)d_in[0];
    const int*   ei   = (const int*)d_in[1];
    const float* w_l  = (const float*)d_in[2];
    const float* b_l  = (const float*)d_in[3];
    const float* w_r  = (const float*)d_in[4];
    const float* b_r  = (const float*)d_in[5];
    const float* att  = (const float*)d_in[6];
    const float* bias = (const float*)d_in[7];
    const float* ln_w = (const float*)d_in[8];
    const float* ln_b = (const float*)d_in[9];
    float* out = (float*)d_out;

    const int N  = in_sizes[0] / D_IN;
    const int E  = in_sizes[1] / 2;
    const int Et = E + N;
    const int total = N * HC;

    // workspace carve-up (all region sizes are multiples of 16 bytes)
    char* ws = (char*)d_ws;
    _Float16* w16    = (_Float16*)ws;                        // 2*128*128*2 B = 64 KB
    float*    xl     = (float*)(ws + 2 * D_IN * HC * sizeof(_Float16));
    float*    xr     = xl + (size_t)N * HC;
    float*    logits = xr + (size_t)N * HC;                  // Et*4 floats (logit, then exp)
    unsigned* mmax   = (unsigned*)(logits + (size_t)Et * NHEAD);
    float*    denom  = (float*)(mmax + (size_t)N * NHEAD);
    double*   acc    = (double*)(denom + (size_t)N * NHEAD);
    float*    mi     = (float*)(acc + 2);

    // 0. weights -> f16
    k_convert_w<<<(D_IN * HC + 255) / 256, 256, 0, stream>>>(w_l, w_r, w16);
    // 1. WMMA GEMMs (LDS-staged A, register-resident B)
    k_gemm<<<(N + BM - 1) / BM, 512, 0, stream>>>(x, w16, b_l, b_r, xl, xr, N);
    // 2. init accumulators / output
    k_init<<<(total + 255) / 256, 256, 0, stream>>>(out, total, denom, mmax, N * NHEAD, acc);
    // 3. logits + segment max
    k_edge_logits<<<(Et + 7) / 8, 256, 0, stream>>>(ei, E, N, xl, xr, att, logits, mmax);
    // 4. exp + segment sum
    k_edge_exp<<<(Et * NHEAD + 255) / 256, 256, 0, stream>>>(ei, E, N, mmax, logits, denom);
    // 5. alpha-weighted aggregation
    k_edge_agg<<<(Et + 7) / 8, 256, 0, stream>>>(ei, E, N, xl, logits, denom, out);
    // 6. bias + global stats
    k_bias_stats<<<(total + 255) / 256, 256, 0, stream>>>(out, bias, total, acc);
    // 7. finalize mean / inv-std
    k_finalize<<<1, 1, 0, stream>>>(acc, mi, (double)total);
    // 8. apply LN affine
    k_apply<<<(total + 255) / 256, 256, 0, stream>>>(out, mi, ln_w, ln_b, total);
}
